// StableSSM_8203387536008
// MI455X (gfx1250) — matrix operand order repaired and checked by reading, not compiled, not run on hardware
//
#include <hip/hip_runtime.h>

typedef unsigned short u16;
typedef __attribute__((ext_vector_type(16))) __bf16 v16bf;
typedef __attribute__((ext_vector_type(8)))  float  v8f;

#define DEVINL __device__ __forceinline__

DEVINL u16 f2bf(float x) {
    unsigned int b = __float_as_uint(x);
    unsigned int r = b + 0x7FFFu + ((b >> 16) & 1u);   // round-to-nearest-even
    return (u16)(r >> 16);
}
DEVINL float bf2f(u16 u) { return __uint_as_float(((unsigned int)u) << 16); }

union BF16Frag { v16bf v; u16 u[16]; };

DEVINL v8f wmma_bf16(const BF16Frag& a, const BF16Frag& b, v8f c) {
    return __builtin_amdgcn_wmma_f32_16x16x32_bf16(
        false, a.v, false, b.v, (short)0, c, false, false);
}

// A-fragment (16x32 bf16, M=row, K): lane holds row (l&15); p is offset by kA=(half*8)
// elements j=0..15 -> K = half*16 + j (contiguous, matches B)
DEVINL void load_afrag(BF16Frag& f, const u16* p /* = row_base + k0 + kA */) {
    const int hh = (threadIdx.x & 31) >> 4;
#pragma unroll
    for (int j = 0; j < 16; ++j) f.u[j] = p[hh * 8 + j];
}
// B-fragment (32x16 bf16, K x N): lane holds col (l&15); K = kB + j, kB=(half*16)
DEVINL void load_bfrag(BF16Frag& f, const u16* p /* = col_row_base + k0 + kB */) {
#pragma unroll
    for (int j = 0; j < 16; ++j) f.u[j] = p[j];
}

// ---------------- prep: weights to bf16, re-layout -------------------------
// Aw[h][k]   = bf16(tanh(A_raw[h][k]) * 0.95)                (256x256)
// Wb[h][lag*128+i] = bf16(B[lag][h][i])                      (256x512)
// Wd[o][lag*128+i] = bf16(D[lag][o][i])                      (128x512)
// Wc[o][h]   = bf16(C[o][h])                                 (128x256)
__global__ void k_prep(const float* __restrict__ A_raw, const float* __restrict__ B,
                       const float* __restrict__ C, const float* __restrict__ D,
                       u16* Aw, u16* Wb, u16* Wd, u16* Wc) {
    const int TOT = 65536 + 131072 + 65536 + 32768;
    for (int idx = blockIdx.x * blockDim.x + threadIdx.x; idx < TOT;
         idx += gridDim.x * blockDim.x) {
        if (idx < 65536) {
            Aw[idx] = f2bf(tanhf(A_raw[idx]) * 0.95f);
        } else if (idx < 196608) {
            int t = idx - 65536; int h = t >> 9; int k = t & 511;
            int lag = k >> 7; int i = k & 127;
            Wb[t] = f2bf(B[(lag * 256 + h) * 128 + i]);
        } else if (idx < 262144) {
            int t = idx - 196608; int o = t >> 9; int k = t & 511;
            int lag = k >> 7; int i = k & 127;
            Wd[t] = f2bf(D[(lag * 128 + o) * 128 + i]);
        } else {
            int t = idx - 262144;
            Wc[t] = f2bf(C[t]);
        }
    }
}

// 8 floats -> 8 packed bf16 per thread (b128 in, b128 out)
__global__ void k_xconv(const float4* __restrict__ x, uint4* __restrict__ xbf) {
    int i = blockIdx.x * blockDim.x + threadIdx.x;
    float4 a = x[2 * i], b = x[2 * i + 1];
    uint4 o;
    o.x = (unsigned)f2bf(a.x) | ((unsigned)f2bf(a.y) << 16);
    o.y = (unsigned)f2bf(a.z) | ((unsigned)f2bf(a.w) << 16);
    o.z = (unsigned)f2bf(b.x) | ((unsigned)f2bf(b.y) << 16);
    o.w = (unsigned)f2bf(b.z) | ((unsigned)f2bf(b.w) << 16);
    xbf[i] = o;
}

// ---------------- conv GEMM: control (bf16) + d_term (f32 into y) ----------
// rows M = 16 consecutive t at fixed b; K = 512 (lag-major); control stored [t][b][h]
__global__ __launch_bounds__(128) void k_conv(const u16* __restrict__ xbf,
                                              const u16* __restrict__ Wb,
                                              const u16* __restrict__ Wd,
                                              u16* __restrict__ control,
                                              float* __restrict__ y) {
    const int b = blockIdx.x;        // 0..15
    const int t0 = blockIdx.y * 16;  // 0..8176
    const int w = threadIdx.x >> 5;
    const int lane = threadIdx.x & 31;
    const int half = lane >> 4, ln = lane & 15;
    const int kA = half * 8, kB = half * 16;

    for (int tile = w; tile < 24; tile += 4) {
        const bool isC = tile < 16;
        const int n0 = isC ? tile * 16 : (tile - 16) * 16;
        const u16* W = isC ? Wb : Wd;
        v8f acc = {0.f, 0.f, 0.f, 0.f, 0.f, 0.f, 0.f, 0.f};
#pragma unroll
        for (int kc = 0; kc < 16; ++kc) {
            const int k0 = kc * 32;
            const int lag = k0 >> 7;
            const int i0 = k0 & 127;
            BF16Frag a;
            const int t = t0 + ln - lag;
            if (t < 0) {
#pragma unroll
                for (int j = 0; j < 16; ++j) a.u[j] = 0;
            } else {
                load_afrag(a, xbf + ((size_t)b * 8192 + t) * 128 + i0 + kA);
            }
            BF16Frag bb;
            load_bfrag(bb, W + (n0 + ln) * 512 + k0 + kB);
            acc = wmma_bf16(a, bb, acc);
        }
        if (isC) {
#pragma unroll
            for (int r = 0; r < 8; ++r) {
                int t = t0 + r + half * 8;
                control[((size_t)t * 16 + b) * 256 + n0 + ln] = f2bf(acc[r]);
            }
        } else {
#pragma unroll
            for (int r = 0; r < 8; ++r) {
                int t = t0 + r + half * 8;
                y[((size_t)b * 8192 + t) * 128 + n0 + ln] = acc[r];
            }
        }
    }
}

// ---------------- 256x256 bf16 squaring: Pout = Pin @ Pin ------------------
__global__ __launch_bounds__(512) void k_sq(const u16* __restrict__ Pin,
                                            u16* __restrict__ Pout) {
    const int w = threadIdx.x >> 5;
    const int lane = threadIdx.x & 31;
    const int half = lane >> 4, ln = lane & 15;
    const int kA = half * 8, kB = half * 16;
    const int tile = blockIdx.x * 16 + w;          // 256 tiles
    const int r0 = (tile >> 4) * 16, c0 = (tile & 15) * 16;
    v8f acc = {0.f, 0.f, 0.f, 0.f, 0.f, 0.f, 0.f, 0.f};
#pragma unroll
    for (int kc = 0; kc < 8; ++kc) {
        const int k0 = kc * 32;
        BF16Frag a, bb;
        load_afrag(a, Pin + (r0 + ln) * 256 + k0 + kA);
#pragma unroll
        for (int j = 0; j < 16; ++j) bb.u[j] = Pin[(k0 + kB + j) * 256 + c0 + ln];
        acc = wmma_bf16(a, bb, acc);
    }
#pragma unroll
    for (int r = 0; r < 8; ++r)
        Pout[(r0 + r + half * 8) * 256 + c0 + ln] = f2bf(acc[r]);
}

// ---------------- chunk scan (pass1: fg only; pass3: true states) ----------
// One block per chunk g (Q=64 steps). 16 waves, wave w owns h-tile n0=w*16.
// Ping-pong LDS state -> single barrier per step; control prefetched one step ahead.
__global__ __launch_bounds__(512) void k_scan(const u16* __restrict__ Aw,
                                              const u16* __restrict__ control,
                                              const u16* __restrict__ carry,
                                              u16* __restrict__ states,
                                              float* __restrict__ fg,
                                              int writeStates) {
    __shared__ u16 st[2][4096];
    const int g = blockIdx.x;
    const int w = threadIdx.x >> 5;
    const int lane = threadIdx.x & 31;
    const int half = lane >> 4, ln = lane & 15;
    const int kA = half * 8, kB = half * 16;
    const int n0 = w * 16;

    // resident B-fragments of A^T (Aw row-major [h][k]) for this wave's tile
    BF16Frag bfr[8];
#pragma unroll
    for (int kc = 0; kc < 8; ++kc)
        load_bfrag(bfr[kc], Aw + (n0 + ln) * 256 + kc * 32 + kB);

    if (writeStates) {
        for (int i = threadIdx.x; i < 4096; i += 512) st[0][i] = carry[g * 4096 + i];
    } else {
        for (int i = threadIdx.x; i < 4096; i += 512) st[0][i] = 0;
    }

    const int coff = (half * 8) * 256 + n0 + ln;   // + r*256 per accumulator row
    float creg[8];
    {
        const u16* cp = control + (size_t)(g * 64) * 4096;
#pragma unroll
        for (int r = 0; r < 8; ++r) creg[r] = bf2f(cp[coff + r * 256]);
    }
    __syncthreads();

    int p = 0;
    v8f acc = {0.f, 0.f, 0.f, 0.f, 0.f, 0.f, 0.f, 0.f};
    for (int tl = 0; tl < 64; ++tl) {
        const int t = g * 64 + tl;
        BF16Frag af[8];
#pragma unroll
        for (int kc = 0; kc < 8; ++kc)
            load_afrag(af[kc], st[p] + ln * 256 + kc * 32 + kA);
        float cnext[8];
        if (tl < 63) {                 // prefetch next step's control (global)
            const u16* cp = control + (size_t)(t + 1) * 4096;
#pragma unroll
            for (int r = 0; r < 8; ++r) cnext[r] = bf2f(cp[coff + r * 256]);
        }
#pragma unroll
        for (int r = 0; r < 8; ++r) acc[r] = creg[r];
#pragma unroll
        for (int kc = 0; kc < 8; ++kc) acc = wmma_bf16(af[kc], bfr[kc], acc);
#pragma unroll
        for (int r = 0; r < 8; ++r) {
            const int b = r + half * 8;
            const u16 v = f2bf(acc[r]);
            st[p ^ 1][b * 256 + n0 + ln] = v;
            if (writeStates) states[((size_t)t * 16 + b) * 256 + n0 + ln] = v;
        }
        __syncthreads();
        if (tl < 63) {
#pragma unroll
            for (int r = 0; r < 8; ++r) creg[r] = cnext[r];
        }
        p ^= 1;
    }
    if (!writeStates) {
#pragma unroll
        for (int r = 0; r < 8; ++r)
            fg[((size_t)g * 16 + r + half * 8) * 256 + n0 + ln] = acc[r];
    }
}

// ---------------- sequential chunk combine: carry_{g+1}=carry_g A^Q^T + f_g
__global__ __launch_bounds__(512) void k_combine(const u16* __restrict__ A64,
                                                 const float* __restrict__ fg,
                                                 u16* __restrict__ carry) {
    __shared__ u16 st[2][4096];
    const int w = threadIdx.x >> 5;
    const int lane = threadIdx.x & 31;
    const int half = lane >> 4, ln = lane & 15;
    const int kA = half * 8, kB = half * 16;
    const int n0 = w * 16;

    BF16Frag bfr[8];
#pragma unroll
    for (int kc = 0; kc < 8; ++kc)
        load_bfrag(bfr[kc], A64 + (n0 + ln) * 256 + kc * 32 + kB);
    for (int i = threadIdx.x; i < 4096; i += 512) st[0][i] = 0;

    const int foff = (half * 8) * 256 + n0 + ln;
    float creg[8];
#pragma unroll
    for (int r = 0; r < 8; ++r) creg[r] = fg[foff + r * 256];
    __syncthreads();

    int p = 0;
    for (int g = 0; g < 128; ++g) {
        for (int i = threadIdx.x; i < 4096; i += 512) carry[g * 4096 + i] = st[p][i];
        BF16Frag af[8];
#pragma unroll
        for (int kc = 0; kc < 8; ++kc)
            load_afrag(af[kc], st[p] + ln * 256 + kc * 32 + kA);
        float cnext[8];
        if (g < 127) {                 // prefetch next chunk's fg
            const float* fp = fg + (size_t)(g + 1) * 4096;
#pragma unroll
            for (int r = 0; r < 8; ++r) cnext[r] = fp[foff + r * 256];
        }
        v8f acc;
#pragma unroll
        for (int r = 0; r < 8; ++r) acc[r] = creg[r];
#pragma unroll
        for (int kc = 0; kc < 8; ++kc) acc = wmma_bf16(af[kc], bfr[kc], acc);
#pragma unroll
        for (int r = 0; r < 8; ++r)
            st[p ^ 1][(r + half * 8) * 256 + n0 + ln] = f2bf(acc[r]);
        __syncthreads();
        if (g < 127) {
#pragma unroll
            for (int r = 0; r < 8; ++r) creg[r] = cnext[r];
        }
        p ^= 1;
    }
}

// ---------------- output GEMM: y += states @ C^T ---------------------------
__global__ __launch_bounds__(128) void k_out(const u16* __restrict__ states,
                                             const u16* __restrict__ Wc,
                                             float* __restrict__ y) {
    const int b = blockIdx.x;
    const int t0 = blockIdx.y * 16;
    const int w = threadIdx.x >> 5;
    const int lane = threadIdx.x & 31;
    const int half = lane >> 4, ln = lane & 15;
    const int kA = half * 8, kB = half * 16;

    for (int tile = w; tile < 8; tile += 4) {
        const int n0 = tile * 16;
        v8f acc = {0.f, 0.f, 0.f, 0.f, 0.f, 0.f, 0.f, 0.f};
#pragma unroll
        for (int kc = 0; kc < 8; ++kc) {
            const int k0 = kc * 32;
            BF16Frag a, bb;
            load_afrag(a, states + ((size_t)(t0 + ln) * 16 + b) * 256 + k0 + kA);
            load_bfrag(bb, Wc + (n0 + ln) * 256 + k0 + kB);
            acc = wmma_bf16(a, bb, acc);
        }
#pragma unroll
        for (int r = 0; r < 8; ++r) {
            int t = t0 + r + half * 8;
            size_t yi = ((size_t)b * 8192 + t) * 128 + n0 + ln;
            y[yi] += acc[r];
        }
    }
}

extern "C" void kernel_launch(void* const* d_in, const int* in_sizes, int n_in,
                              void* d_out, int out_size, void* d_ws, size_t ws_size,
                              hipStream_t stream) {
    const float* x     = (const float*)d_in[0];  // [16,8192,128]
    const float* A_raw = (const float*)d_in[1];  // [256,256]
    const float* B     = (const float*)d_in[2];  // [4,256,128]
    const float* C     = (const float*)d_in[3];  // [128,256]
    const float* D     = (const float*)d_in[4];  // [4,128,128]
    float* y = (float*)d_out;                    // [16,8192,128]

    char* ws = (char*)d_ws;
    u16*  xbf     = (u16*)(ws);                       //  33,554,432 B
    u16*  control = (u16*)(ws + 33554432ull);         //  67,108,864 B
    u16*  states  = (u16*)(ws + 100663296ull);        //  67,108,864 B
    u16*  Aw      = (u16*)(ws + 167772160ull);        //     131,072 B
    u16*  Pa      = (u16*)(ws + 167903232ull);        //     131,072 B
    u16*  Pb      = (u16*)(ws + 168034304ull);        //     131,072 B
    u16*  Wb      = (u16*)(ws + 168165376ull);        //     262,144 B
    u16*  Wd      = (u16*)(ws + 168427520ull);        //     131,072 B
    u16*  Wc      = (u16*)(ws + 168558592ull);        //      65,536 B
    float* fg     = (float*)(ws + 168624128ull);      //   2,097,152 B
    u16*  carry   = (u16*)(ws + 170721280ull);        //   1,048,576 B

    k_prep<<<1152, 256, 0, stream>>>(A_raw, B, C, D, Aw, Wb, Wd, Wc);
    k_xconv<<<8192, 256, 0, stream>>>((const float4*)x, (uint4*)xbf);
    k_conv<<<dim3(16, 512), 128, 0, stream>>>(xbf, Wb, Wd, control, y);

    // A^64 by log-doubling
    k_sq<<<16, 512, 0, stream>>>(Aw, Pa);   // A^2
    k_sq<<<16, 512, 0, stream>>>(Pa, Pb);   // A^4
    k_sq<<<16, 512, 0, stream>>>(Pb, Pa);   // A^8
    k_sq<<<16, 512, 0, stream>>>(Pa, Pb);   // A^16
    k_sq<<<16, 512, 0, stream>>>(Pb, Pa);   // A^32
    k_sq<<<16, 512, 0, stream>>>(Pa, Pb);   // A^64

    k_scan<<<128, 512, 0, stream>>>(Aw, control, carry, states, fg, 0);
    k_combine<<<1, 512, 0, stream>>>(Pb, fg, carry);
    k_scan<<<128, 512, 0, stream>>>(Aw, control, carry, states, fg, 1);
    k_out<<<dim3(16, 512), 128, 0, stream>>>(states, Wc, y);
}
